// DeformConv2d_44418551775890
// MI455X (gfx1250) — matrix operand order repaired and checked by reading, not compiled, not run on hardware
//
#include <hip/hip_runtime.h>

typedef __attribute__((ext_vector_type(16))) _Float16     v16h;
typedef __attribute__((ext_vector_type(8)))  float        v8f;
typedef __attribute__((ext_vector_type(4)))  unsigned int v4u;

#define HWSZ 4096
#define HH   64
#define WW   64
#define CIN  128
#define COUT 256
#define KKT  9
#define KDIM (CIN * KKT) /* 1152 */
#define BATCH 4

// Async memory->LDS path (CDNA5): guarded so we always compile.
#if defined(__gfx1250__) && __has_builtin(__builtin_amdgcn_global_load_async_to_lds_b128) && __has_builtin(__builtin_amdgcn_s_wait_asynccnt)
#define USE_ASYNC_LDS 1
typedef int v4i_ __attribute__((__vector_size__(16)));
typedef __attribute__((address_space(1))) v4i_ g_v4i;   // global int4
typedef __attribute__((address_space(3))) v4i_ l_v4i;   // LDS int4
#else
#define USE_ASYNC_LDS 0
#endif

union Frag { v16h h; v4u u[2]; };

// ---------------- kernel 0: weight -> f16, layout [o][c*9+k] (identity) ----
__global__ __launch_bounds__(256) void wcvt_kernel(const float* __restrict__ w,
                                                   _Float16* __restrict__ wf) {
    int i = blockIdx.x * 256 + threadIdx.x;
    if (i < COUT * KDIM) wf[i] = (_Float16)w[i];
}

// ---------------- kernel 1: offset/mask 3x3 conv -> om[b][27][hw] ----------
__global__ __launch_bounds__(256) void omconv_kernel(const float* __restrict__ x,
                                                     const float* __restrict__ ow,
                                                     const float* __restrict__ ob,
                                                     float* __restrict__ om) {
    int idx = blockIdx.x * 256 + threadIdx.x;   // 27*B*HW threads exactly
    int oc  = idx / (BATCH * HWSZ);
    int rem = idx - oc * (BATCH * HWSZ);
    int b   = rem >> 12;
    int hw  = rem & (HWSZ - 1);
    int py  = hw >> 6, px = hw & 63;

    float acc = ob[oc];
    const float* wrow = ow + (size_t)oc * KDIM;
    for (int c = 0; c < CIN; ++c) {
        const float* xc = x + ((size_t)b * CIN + c) * HWSZ;
        const float* wc = wrow + c * 9;
#pragma unroll
        for (int ky = 0; ky < 3; ++ky) {
            int y = py - 1 + ky;
            if ((unsigned)y < 64u) {
#pragma unroll
                for (int kx = 0; kx < 3; ++kx) {
                    int xx = px - 1 + kx;
                    if ((unsigned)xx < 64u) acc += xc[y * 64 + xx] * wc[ky * 3 + kx];
                }
            }
        }
    }
    om[((size_t)b * 27 + oc) * HWSZ + hw] = acc;
}

// ---------------- kernel 2: deformable bilinear sampling -> f16 ------------
// smp layout: [b][hw][kdim] with kdim = c*9+k  (rows contiguous in K for GEMM)
__global__ __launch_bounds__(256) void sample_kernel(const float* __restrict__ x,
                                                     const float* __restrict__ om,
                                                     _Float16* __restrict__ smp) {
    int idx  = blockIdx.x * 256 + threadIdx.x;  // B*HW*16 threads exactly
    int cg   = idx & 15;                        // channel group (8 ch each)
    int rest = idx >> 4;
    int hw   = rest & (HWSZ - 1);
    int b    = rest >> 12;
    int py = hw >> 6, px = hw & 63;

    const float* omb = om + (size_t)b * 27 * HWSZ + hw;
    __align__(16) _Float16 buf[72];
    const float SC = 64.0f / 63.0f;  // W/(W-1)

#pragma unroll
    for (int k = 0; k < 9; ++k) {
        float offx = omb[(size_t)k * HWSZ];
        float offy = omb[(size_t)(9 + k) * HWSZ];
        float mz   = omb[(size_t)(18 + k) * HWSZ];
        float mk   = 1.0f / (1.0f + __expf(-mz));
        int kox = (k % 3) - 1, koy = (k / 3) - 1;
        float sx = (float)(px - 1 + kox) + offx;
        float sy = (float)(py - 1 + koy) + offy;
        float ix = sx * SC - 0.5f;
        float iy = sy * SC - 0.5f;
        float x0f = floorf(ix), y0f = floorf(iy);
        float wx1 = ix - x0f, wx0 = 1.0f - wx1;
        float wy1 = iy - y0f, wy0 = 1.0f - wy1;
        int xi0 = (int)x0f, yi0 = (int)y0f;
        int xi1 = xi0 + 1,  yi1 = yi0 + 1;
        bool vx0 = (unsigned)xi0 < 64u, vx1 = (unsigned)xi1 < 64u;
        bool vy0 = (unsigned)yi0 < 64u, vy1 = (unsigned)yi1 < 64u;
        int cx0 = min(max(xi0, 0), 63), cx1 = min(max(xi1, 0), 63);
        int cy0 = min(max(yi0, 0), 63), cy1 = min(max(yi1, 0), 63);
        float w00 = (vy0 && vx0) ? wy0 * wx0 : 0.0f;
        float w01 = (vy0 && vx1) ? wy0 * wx1 : 0.0f;
        float w10 = (vy1 && vx0) ? wy1 * wx0 : 0.0f;
        float w11 = (vy1 && vx1) ? wy1 * wx1 : 0.0f;
        int i00 = cy0 * 64 + cx0, i01 = cy0 * 64 + cx1;
        int i10 = cy1 * 64 + cx0, i11 = cy1 * 64 + cx1;
#pragma unroll
        for (int c = 0; c < 8; ++c) {
            const float* xb = x + ((size_t)b * CIN + cg * 8 + c) * HWSZ;
            float v = w00 * xb[i00] + w01 * xb[i01] + w10 * xb[i10] + w11 * xb[i11];
            buf[c * 9 + k] = (_Float16)(v * mk);
        }
    }
    v4u* dst = (v4u*)(smp + ((size_t)b * HWSZ + hw) * (size_t)KDIM + cg * 72);
#pragma unroll
    for (int r2 = 0; r2 < 9; ++r2) dst[r2] = ((const v4u*)buf)[r2];
}

// ---------------- kernel 3: WMMA GEMM  out[b][o][hw] = A(256xK) * B(KxN) ---
// block = 256 threads = 8 wave32; tile 128(M) x 128(N); K-step 32 (f16 WMMA)
// Async path: double-buffered LDS, GLOBAL_LOAD_ASYNC_TO_LDS_B128, one barrier
// per K-step, loads for tile i+1 overlap WMMA of tile i.
__global__ __launch_bounds__(256) void gemm_kernel(const _Float16* __restrict__ Af,
                                                   const _Float16* __restrict__ Bf,
                                                   const float* __restrict__ bias,
                                                   float* __restrict__ out) {
#if USE_ASYNC_LDS
    __shared__ v4u ldsA[2][128 * 5];   // rows padded to 80B to spread banks
    __shared__ v4u ldsB[2][128 * 5];
#else
    __shared__ v4u ldsA[1][128 * 5];
    __shared__ v4u ldsB[1][128 * 5];
#endif

    const int tid  = threadIdx.x;
    const int lane = tid & 31;
    const int wv   = tid >> 5;
    const int wvM  = wv & 1;    // 2 waves along M (64 rows each)
    const int wvN  = wv >> 1;   // 4 waves along N (32 cols each)
    const int ntile = blockIdx.x;   // 32
    const int mtile = blockIdx.y;   // 2
    const int b     = blockIdx.z;   // 4

    const int r     = tid >> 1;     // 0..127: row of A-tile / n-row of B-tile
    const int chunk = tid & 1;      // two 16B chunks of the 32B half-row
    const int ldsIdx = r * 5 + chunk * 2;

    const _Float16* Ag = Af + (size_t)(mtile * 128 + r) * KDIM;
    const _Float16* Bg = Bf + ((size_t)b * HWSZ + ntile * 128 + r) * (size_t)KDIM;

    v8f zero = {};
    v8f acc[4][2];
#pragma unroll
    for (int fi = 0; fi < 4; ++fi)
#pragma unroll
        for (int fj = 0; fj < 2; ++fj) acc[fi][fj] = zero;

    const int mrow = lane & 15;
    const int sel  = lane >> 4;

#if USE_ASYNC_LDS
    auto issue = [&](int buf, int k0) {
        const v4u* sA = (const v4u*)(Ag + k0) + chunk * 2;
        const v4u* sB = (const v4u*)(Bg + k0) + chunk * 2;
        __builtin_amdgcn_global_load_async_to_lds_b128(
            (g_v4i*)(sA),     (l_v4i*)(&ldsA[buf][ldsIdx]),     0, 0);
        __builtin_amdgcn_global_load_async_to_lds_b128(
            (g_v4i*)(sA + 1), (l_v4i*)(&ldsA[buf][ldsIdx + 1]), 0, 0);
        __builtin_amdgcn_global_load_async_to_lds_b128(
            (g_v4i*)(sB),     (l_v4i*)(&ldsB[buf][ldsIdx]),     0, 0);
        __builtin_amdgcn_global_load_async_to_lds_b128(
            (g_v4i*)(sB + 1), (l_v4i*)(&ldsB[buf][ldsIdx + 1]), 0, 0);
    };
    issue(0, 0);
    for (int it = 0; it < KDIM / 32; ++it) {
        const int k0  = it * 32;
        const int cur = it & 1;
        __builtin_amdgcn_s_wait_asynccnt(0);  // my tile-`cur` loads landed
        __syncthreads();                      // everyone's landed; prev reads done
        if (k0 + 32 < KDIM) issue(cur ^ 1, k0 + 32);  // overlap with compute

        Frag a[4];
#pragma unroll
        for (int fi = 0; fi < 4; ++fi) {
            int row = wvM * 64 + fi * 16 + mrow;
            a[fi].u[0] = ldsA[cur][row * 5 + sel * 2];
            a[fi].u[1] = ldsA[cur][row * 5 + sel * 2 + 1];
        }
        Frag bb[2];
#pragma unroll
        for (int fj = 0; fj < 2; ++fj) {
            int col = wvN * 32 + fj * 16 + mrow;
            bb[fj].u[0] = ldsB[cur][col * 5 + sel * 2];
            bb[fj].u[1] = ldsB[cur][col * 5 + sel * 2 + 1];
        }
#pragma unroll
        for (int fi = 0; fi < 4; ++fi)
#pragma unroll
            for (int fj = 0; fj < 2; ++fj)
                acc[fi][fj] = __builtin_amdgcn_wmma_f32_16x16x32_f16(
                    false, a[fi].h, false, bb[fj].h, (short)0, acc[fi][fj],
                    false, false);
    }
#else
    for (int k0 = 0; k0 < KDIM; k0 += 32) {
        const v4u* sA = (const v4u*)(Ag + k0) + chunk * 2;
        const v4u* sB = (const v4u*)(Bg + k0) + chunk * 2;
        v4u a0 = sA[0], a1 = sA[1];
        v4u b0 = sB[0], b1 = sB[1];
        if (k0 + 32 < KDIM) {
            __builtin_prefetch((const void*)(Ag + k0 + 32), 0, 3);
            __builtin_prefetch((const void*)(Bg + k0 + 32), 0, 3);
        }
        __syncthreads();
        ldsA[0][ldsIdx]     = a0;
        ldsA[0][ldsIdx + 1] = a1;
        ldsB[0][ldsIdx]     = b0;
        ldsB[0][ldsIdx + 1] = b1;
        __syncthreads();

        Frag a[4];
#pragma unroll
        for (int fi = 0; fi < 4; ++fi) {
            int row = wvM * 64 + fi * 16 + mrow;
            a[fi].u[0] = ldsA[0][row * 5 + sel * 2];
            a[fi].u[1] = ldsA[0][row * 5 + sel * 2 + 1];
        }
        Frag bb[2];
#pragma unroll
        for (int fj = 0; fj < 2; ++fj) {
            int col = wvN * 32 + fj * 16 + mrow;
            bb[fj].u[0] = ldsB[0][col * 5 + sel * 2];
            bb[fj].u[1] = ldsB[0][col * 5 + sel * 2 + 1];
        }
#pragma unroll
        for (int fi = 0; fi < 4; ++fi)
#pragma unroll
            for (int fj = 0; fj < 2; ++fj)
                acc[fi][fj] = __builtin_amdgcn_wmma_f32_16x16x32_f16(
                    false, a[fi].h, false, bb[fj].h, (short)0, acc[fi][fj],
                    false, false);
    }
#endif

    // epilogue: C/D layout -> VGPR e: M = base + 8*sel + e, N = lane&15
#pragma unroll
    for (int fi = 0; fi < 4; ++fi) {
#pragma unroll
        for (int fj = 0; fj < 2; ++fj) {
            int nIdx = ntile * 128 + wvN * 32 + fj * 16 + (lane & 15);
#pragma unroll
            for (int e = 0; e < 8; ++e) {
                int m = mtile * 128 + wvM * 64 + fi * 16 + sel * 8 + e;
                out[((size_t)b * COUT + m) * HWSZ + nIdx] = acc[fi][fj][e] + bias[m];
            }
        }
    }
}

extern "C" void kernel_launch(void* const* d_in, const int* in_sizes, int n_in,
                              void* d_out, int out_size, void* d_ws, size_t ws_size,
                              hipStream_t stream) {
    const float* x    = (const float*)d_in[0];
    const float* om_w = (const float*)d_in[1];
    const float* om_b = (const float*)d_in[2];
    const float* wgt  = (const float*)d_in[3];
    const float* bias = (const float*)d_in[4];
    float* out = (float*)d_out;

    char* ws = (char*)d_ws;
    _Float16* wf16 = (_Float16*)ws;                            // 589,824 B
    float*    om   = (float*)(ws + 589824);                    // 1,769,472 B
    _Float16* smp  = (_Float16*)(ws + 589824 + 1769472);       // 37,748,736 B

    wcvt_kernel  <<<1152, 256, 0, stream>>>(wgt, wf16);
    omconv_kernel<<<1728, 256, 0, stream>>>(x, om_w, om_b, om);
    sample_kernel<<<1024, 256, 0, stream>>>(x, om, smp);
    gemm_kernel  <<<dim3(32, 2, 4), 256, 0, stream>>>(wf16, smp, bias, out);
}
